// SaintEncoderDecoderModel_8065948582460
// MI455X (gfx1250) — compile-verified, matrix operand order checked by reading, not run on hardware
//
#include <hip/hip_runtime.h>
#include <hip/hip_bf16.h>
#include <math.h>

#define B_   16
#define S_   512
#define D_   512
#define H_   8
#define DH_  64
#define F_   2048
#define E_   2048
#define RE_  128
#define L_   4
#define MROWS (B_*S_)          /* 8192 */
#define ENC_K (2*E_)           /* 4096 */
#define DEC_K (RE_+3*E_)       /* 6272 */

typedef __attribute__((ext_vector_type(16))) __bf16 v16bf;
typedef __attribute__((ext_vector_type(8)))  float  v8f;

union Frag { v16bf v; unsigned int u[8]; };

__device__ __forceinline__ unsigned short f2bf(float f) {
  unsigned int x = __float_as_uint(f);
  x += 0x7fffu + ((x >> 16) & 1u);
  return (unsigned short)(x >> 16);
}

// k-pair base for bf16 A/B fragment vgpr j (ISA 16-bit 16x32 layout)
__device__ __forceinline__ int kpair(int j, int half) {
  return ((j < 4) ? (2 * j) : (16 + 2 * (j - 4))) + 8 * half;
}

// CDNA5 async global->LDS copy (ASYNCcnt), GV addressing mode.
// lds_off: low 32 bits of the generic LDS pointer == LDS byte offset.
__device__ __forceinline__ void async_b128(unsigned lds_off, const void* gaddr) {
  asm volatile("global_load_async_to_lds_b128 %0, %1, off"
               :: "v"(lds_off), "v"(gaddr) : "memory");
}
__device__ __forceinline__ void wait_async0() {
  asm volatile("s_wait_asynccnt 0x0" ::: "memory");
}
__device__ __forceinline__ unsigned lds_off32(const void* p) {
  return (unsigned)(size_t)p;
}

// ---------------------------------------------------------------------------
// Weight convert+swizzle: f32 [K][N] row-major -> bf16 fragment-order tiles.
// Layout: [K/32][N/16] tiles of 512 bf16 (1KB); within a tile, lane l's
// 8 B-fragment uints are 32 contiguous bytes: u[j] = (W[k0][n], W[k0+1][n])
// with k0 = kpair(j, l>>4), n = l&15.  One thread per packed uint.
// ---------------------------------------------------------------------------
__global__ void swz_w_bf16(const float* __restrict__ src, unsigned int* __restrict__ dst,
                           int K, int N)
{
  size_t idx = (size_t)blockIdx.x * 256 + threadIdx.x;
  size_t total = (size_t)K * N / 2;
  if (idx >= total) return;
  int j    = (int)(idx & 7);
  int lane = (int)((idx >> 3) & 31);
  size_t tile = idx >> 8;
  int ntn  = N >> 4;
  int tileN = (int)(tile % ntn);
  int tileK = (int)(tile / ntn);
  int half = lane >> 4, l15 = lane & 15;
  int k0 = tileK * 32 + kpair(j, half);
  int n  = tileN * 16 + l15;
  unsigned lo = f2bf(src[(size_t)k0 * N + n]);
  unsigned hi = f2bf(src[(size_t)(k0 + 1) * N + n]);
  dst[idx] = lo | (hi << 16);
}

// ---------------------------------------------------------------------------
// Generic bf16 GEMM:  C[M,N](f32) = A[M,K](bf16) * W[K,N](bf16,swizzled) + bias
// optional residual add, exact GELU, f32 and/or bf16 output.
// Block tile 128x64, 8 waves of 32x32 (2x2 WMMA 16x16x32 bf16), K-step 32.
// Double-buffered LDS fed by global_load_async_to_lds_b128: tile t+1 copies
// are issued before tile t's WMMAs, so ASYNCcnt latency hides under math.
// B arrives pre-swizzled -> staging is one contiguous async b128 per thread
// and B-fragment reads are two ds_load_b128 per fragment (same as A).
// All dims are exact tile multiples -> no bounds checks, EXEC all ones.
// ---------------------------------------------------------------------------
__global__ __launch_bounds__(256) void gemm_bf16(
    const unsigned short* __restrict__ A, int lda,
    const unsigned short* __restrict__ Wsw,
    const float* __restrict__ bias,
    const float* __restrict__ resid,
    float* __restrict__ outF,
    unsigned short* __restrict__ outB,
    int M, int N, int K, int act)
{
  __shared__ __align__(16) unsigned short As[2][128][40]; // [m][k], 80B row (16B mult)
  __shared__ __align__(16) unsigned short Bs[2][2048];    // 4 fragment tiles of 1KB

  const int tid  = threadIdx.x;
  const int lane = tid & 31;
  const int wid  = tid >> 5;
  const int wm   = wid >> 1;          // 0..3
  const int wn   = wid & 1;           // 0..1
  const int half = lane >> 4;
  const int l15  = lane & 15;
  const int m0   = blockIdx.y * 128;
  const int n0   = blockIdx.x * 64;

  const int ar = tid >> 1;            // A row covered by this thread
  const int ac = (tid & 1) * 16;      // A col base (16 elems = 2 x b128)

  auto stage = [&](int tile, int buf) {
    const unsigned short* srcA = A + (size_t)(m0 + ar) * lda + tile * 32 + ac;
    unsigned aoff = lds_off32(&As[buf][ar][ac]);
    async_b128(aoff,      srcA);
    async_b128(aoff + 16, srcA + 8);
    const unsigned short* srcB =
        Wsw + ((size_t)tile * (N >> 4) + (n0 >> 4)) * 512 + tid * 8;
    async_b128(lds_off32(&Bs[buf][tid * 8]), srcB);
  };

  v8f acc[2][2] = {};
  const int ntiles = K / 32;

  stage(0, 0);
  wait_async0();
  __syncthreads();

  int cur = 0;
  for (int t = 0; t < ntiles; ++t) {
    if (t + 1 < ntiles) stage(t + 1, cur ^ 1);   // prefetch next tile (async)

    Frag a[2], b[2];
#pragma unroll
    for (int mt = 0; mt < 2; ++mt) {
      int row = wm * 32 + mt * 16 + l15;
#pragma unroll
      for (int j = 0; j < 8; ++j)
        a[mt].u[j] = *(const unsigned int*)&As[cur][row][kpair(j, half)];
    }
#pragma unroll
    for (int nt = 0; nt < 2; ++nt) {
      const unsigned int* bp =
          (const unsigned int*)&Bs[cur][(wn * 2 + nt) * 512 + lane * 16];
#pragma unroll
      for (int j = 0; j < 8; ++j)
        b[nt].u[j] = bp[j];
    }
#pragma unroll
    for (int mt = 0; mt < 2; ++mt)
#pragma unroll
      for (int nt = 0; nt < 2; ++nt)
        acc[mt][nt] = __builtin_amdgcn_wmma_f32_16x16x32_bf16(
            false, a[mt].v, false, b[nt].v, (short)0, acc[mt][nt], false, false);

    wait_async0();
    __syncthreads();
    cur ^= 1;
  }

  // epilogue (C layout: lane<16 -> n=lane, m=vgpr; lane>=16 -> m=vgpr+8)
#pragma unroll
  for (int mt = 0; mt < 2; ++mt) {
#pragma unroll
    for (int nt = 0; nt < 2; ++nt) {
      int ng = n0 + wn * 32 + nt * 16 + l15;
      float bv = bias ? bias[ng] : 0.f;
#pragma unroll
      for (int i = 0; i < 8; ++i) {
        int mg = m0 + wm * 32 + mt * 16 + i + 8 * half;
        float v = acc[mt][nt][i] + bv;
        if (resid) v += resid[(size_t)mg * N + ng];
        if (act == 1) v = 0.5f * v * (1.f + erff(v * 0.70710678118654752f));
        if (outF) outF[(size_t)mg * N + ng] = v;
        if (outB) outB[(size_t)mg * N + ng] = f2bf(v);
      }
    }
  }
}

// ---------------------------------------------------------------------------
// Flash-style attention, one block = (b, h, 64 queries), 4 waves x 16 queries.
// K/V streamed in 32-key chunks via async global->LDS copies; scores + P.V
// via bf16 WMMA, online softmax with 16-lane shfl_xor row reductions.
// Per-element score transform: s = qk / (9 - 1/(max(lag,0)+1)) + pb [+ causal]
// ---------------------------------------------------------------------------
__global__ __launch_bounds__(128) void attn_kernel(
    const unsigned short* __restrict__ qb,
    const unsigned short* __restrict__ kb,
    const unsigned short* __restrict__ vb,
    const float* __restrict__ pbias,     // [H][S][S]
    const int* __restrict__ tsq,         // [B][S]
    const int* __restrict__ tsk,         // [B][S]
    unsigned short* __restrict__ ctx,
    int causal)
{
  __shared__ __align__(16) unsigned short Ks[32][72];    // [key][d0..63] (+pad, 144B row)
  __shared__ __align__(16) unsigned short Vs[32][72];    // [key][d0..63]
  __shared__ __align__(16) unsigned short Ps[4][16][34]; // per-wave P relayout buffer

  const int b    = blockIdx.z;
  const int h    = blockIdx.y;
  const int q0   = blockIdx.x * 64;
  const int tid  = threadIdx.x;
  const int lane = tid & 31;
  const int w    = tid >> 5;
  const int half = lane >> 4;
  const int l15  = lane & 15;
  const int qrow = q0 + w * 16;

  const int skey = tid >> 2;           // staging: key row covered
  const int soff = (tid & 3) * 16;     // staging: d chunk (16 elems = 2 x b128)

  // Q fragments for d=0..63 (two 16x32 A-frags), loaded once
  Frag aq[2];
  {
    const unsigned short* qp = qb + (size_t)(b * S_ + qrow + l15) * D_ + h * DH_;
#pragma unroll
    for (int t = 0; t < 2; ++t)
#pragma unroll
      for (int j = 0; j < 8; ++j)
        aq[t].u[j] = *(const unsigned int*)(qp + kpair(j, half) + t * 32);
  }

  float mrow[8], lrow[8];
  int   tqv[8];
#pragma unroll
  for (int i = 0; i < 8; ++i) {
    mrow[i] = -1e30f; lrow[i] = 0.f;
    tqv[i] = tsq[b * S_ + qrow + i + 8 * half];
  }
  v8f accO[4] = {};

  for (int kc = 0; kc < S_; kc += 32) {
    // async stage of K and V chunks (32 keys x 64 d, bf16)
    {
      const unsigned short* ksrc = kb + (size_t)(b * S_ + kc + skey) * D_ + h * DH_ + soff;
      unsigned koff = lds_off32(&Ks[skey][soff]);
      async_b128(koff,      ksrc);
      async_b128(koff + 16, ksrc + 8);
      const unsigned short* vsrc = vb + (size_t)(b * S_ + kc + skey) * D_ + h * DH_ + soff;
      unsigned voff = lds_off32(&Vs[skey][soff]);
      async_b128(voff,      vsrc);
      async_b128(voff + 16, vsrc + 8);
    }
    wait_async0();
    __syncthreads();

    // scores: S(16x32) = Q(16x64) . K^T, 2 d-steps x 2 key n-tiles
    v8f sc[2] = {};
#pragma unroll
    for (int nt = 0; nt < 2; ++nt) {
#pragma unroll
      for (int t = 0; t < 2; ++t) {
        Frag bk;
#pragma unroll
        for (int j = 0; j < 8; ++j)
          bk.u[j] = *(const unsigned int*)&Ks[nt * 16 + l15][kpair(j, half) + t * 32];
        sc[nt] = __builtin_amdgcn_wmma_f32_16x16x32_bf16(
            false, aq[t].v, false, bk.v, (short)0, sc[nt], false, false);
      }
    }

    // elementwise transform in C layout
    int tkv0 = tsk[b * S_ + kc + l15];
    int tkv1 = tsk[b * S_ + kc + 16 + l15];
    float sval[2][8];
#pragma unroll
    for (int nt = 0; nt < 2; ++nt) {
      int key = kc + nt * 16 + l15;
      int tk  = nt ? tkv1 : tkv0;
#pragma unroll
      for (int i = 0; i < 8; ++i) {
        int qg = qrow + i + 8 * half;
        float lag = (float)(tqv[i] - tk) * (1.f / 60000.f);
        float scl = 9.0f - 1.0f / (fmaxf(lag, 0.f) + 1.f);
        float s = sc[nt][i] / scl + pbias[((size_t)h * S_ + qg) * S_ + key];
        s += (causal && (key > qg)) ? -1e4f : 0.f;
        sval[nt][i] = s;
      }
    }

    // online softmax per row; write P (bf16) to wave-private LDS
#pragma unroll
    for (int i = 0; i < 8; ++i) {
      float mx = fmaxf(sval[0][i], sval[1][i]);
#pragma unroll
      for (int d = 1; d < 16; d <<= 1)
        mx = fmaxf(mx, __shfl_xor(mx, d, 32));
      float nm   = fmaxf(mrow[i], mx);
      float corr = __expf(mrow[i] - nm);
      mrow[i] = nm;
      float e0 = __expf(sval[0][i] - nm);
      float e1 = __expf(sval[1][i] - nm);
      float rs = e0 + e1;
#pragma unroll
      for (int d = 1; d < 16; d <<= 1)
        rs += __shfl_xor(rs, d, 32);
      lrow[i] = lrow[i] * corr + rs;
#pragma unroll
      for (int nt = 0; nt < 4; ++nt)
        accO[nt][i] = accO[nt][i] * corr;
      Ps[w][i + 8 * half][l15]      = f2bf(e0);
      Ps[w][i + 8 * half][16 + l15] = f2bf(e1);
    }
    asm volatile("s_wait_dscnt 0x0" ::: "memory");   // order wave-private P staging

    // P(16x32) . V(32x64): A-frag from Ps, B-frags from Vs (u16 K-pair pack)
    Frag ap;
#pragma unroll
    for (int j = 0; j < 8; ++j)
      ap.u[j] = *(const unsigned int*)&Ps[w][l15][kpair(j, half)];
#pragma unroll
    for (int nt = 0; nt < 4; ++nt) {
      Frag bv;
#pragma unroll
      for (int j = 0; j < 8; ++j) {
        int k0 = kpair(j, half);
        unsigned lo = Vs[k0][nt * 16 + l15];
        unsigned hi = Vs[k0 + 1][nt * 16 + l15];
        bv.u[j] = lo | (hi << 16);
      }
      accO[nt] = __builtin_amdgcn_wmma_f32_16x16x32_bf16(
          false, ap.v, false, bv.v, (short)0, accO[nt], false, false);
    }
    __syncthreads();
  }

  // finalize: ctx = accO / l, bf16, at head column block
#pragma unroll
  for (int nt = 0; nt < 4; ++nt)
#pragma unroll
    for (int i = 0; i < 8; ++i) {
      int m = qrow + i + 8 * half;
      ctx[(size_t)(b * S_ + m) * D_ + h * DH_ + nt * 16 + l15] =
          f2bf(accO[nt][i] / lrow[i]);
    }
}

// ---------------------------------------------------------------------------
// LayerNorm over D=512 (eps=1e-12), writes f32 residual stream + bf16 operand
// ---------------------------------------------------------------------------
__global__ __launch_bounds__(256) void ln_kernel(
    const float* __restrict__ x, const float* __restrict__ g,
    const float* __restrict__ bta, float* __restrict__ outF,
    unsigned short* __restrict__ outB)
{
  __shared__ float red[256];
  const int row = blockIdx.x;
  const int t = threadIdx.x;
  const float* xr = x + (size_t)row * D_;
  float s = 0.f;
  for (int c = t; c < D_; c += 256) s += xr[c];
  red[t] = s; __syncthreads();
  for (int st = 128; st > 0; st >>= 1) { if (t < st) red[t] += red[t + st]; __syncthreads(); }
  float mean = red[0] * (1.f / D_);
  __syncthreads();
  float v = 0.f;
  for (int c = t; c < D_; c += 256) { float d = xr[c] - mean; v += d * d; }
  red[t] = v; __syncthreads();
  for (int st = 128; st > 0; st >>= 1) { if (t < st) red[t] += red[t + st]; __syncthreads(); }
  float rstd = rsqrtf(red[0] * (1.f / D_) + 1e-12f);
  for (int c = t; c < D_; c += 256) {
    float y = (xr[c] - mean) * rstd * g[c] + bta[c];
    if (outF) outF[(size_t)row * D_ + c] = y;
    if (outB) outB[(size_t)row * D_ + c] = f2bf(y);
  }
}

// ---------------------------------------------------------------------------
// prep: lag (per-row previous-distinct-timestamp), ts_new, BN'd numeric
// features, lag/elapsed category buckets
// ---------------------------------------------------------------------------
__global__ void prep_kernel(const int* __restrict__ ts, const float* __restrict__ el,
                            const float* __restrict__ bn_g, const float* __restrict__ bn_b,
                            int* __restrict__ ts_new, float* __restrict__ n0a,
                            float* __restrict__ n1a, int* __restrict__ lagcat,
                            int* __restrict__ elcat)
{
  int idx = blockIdx.x * 256 + threadIdx.x;
  if (idx >= MROWS) return;
  int b = idx / S_, s = idx % S_;
  int t = ts[idx];
  int p = s - 1;
  while (p >= 0 && ts[b * S_ + p] == t) --p;
  float lag = (p >= 0) ? (float)(t - ts[b * S_ + p]) * (1.f / 60000.f) : 0.f;
  lag = fminf(fmaxf(lag, 0.f), 1440.f);
  ts_new[idx] = (s == 0) ? -1 : ts[idx - 1];
  const float binv = 0.999997500009375f;  // rsqrt(1 + 5e-6), eval-mode BN
  n0a[idx] = log1pf(lag) * binv * bn_g[0] + bn_b[0];
  float eln = fminf(fmaxf(el[idx], 0.f), 300.f);
  n1a[idx] = eln * binv * bn_g[1] + bn_b[1];
  lagcat[idx] = (lag < 6.f) ? (int)lag : ((int)((lag - 1.f) / 10.f) + 6);
  int ec = (int)el[idx] + 1;
  elcat[idx] = min(max(ec, 0), 300);
}

__global__ void enc_gather(const int* __restrict__ eids, const int* __restrict__ cids,
                           const float* __restrict__ id_emb, const float* __restrict__ cat_emb,
                           unsigned short* __restrict__ out)
{
  size_t i = (size_t)blockIdx.x * 256 + threadIdx.x;
  if (i >= (size_t)MROWS * ENC_K) return;
  int row = (int)(i / ENC_K);
  int c   = (int)(i % ENC_K);
  float v = (c < E_) ? id_emb[(size_t)eids[row] * E_ + c]
                     : cat_emb[(size_t)cids[row] * E_ + (c - E_)];
  out[i] = f2bf(v);
}

__global__ void dec_gather(const int* __restrict__ rids, const float* __restrict__ resp_emb,
                           const float* __restrict__ num_w, const float* __restrict__ num_b,
                           const float* __restrict__ lag_emb, const float* __restrict__ el_emb,
                           const float* __restrict__ n0a, const float* __restrict__ n1a,
                           const int* __restrict__ lagcat, const int* __restrict__ elcat,
                           unsigned short* __restrict__ out)
{
  size_t i = (size_t)blockIdx.x * 256 + threadIdx.x;
  if (i >= (size_t)MROWS * DEC_K) return;
  int row = (int)(i / DEC_K);
  int c   = (int)(i % DEC_K);
  float v;
  if (c < RE_) {
    v = resp_emb[(size_t)rids[row] * RE_ + c];
  } else if (c < RE_ + E_) {
    int j = c - RE_;
    v = n0a[row] * num_w[j] + n1a[row] * num_w[E_ + j] + num_b[j];
  } else if (c < RE_ + 2 * E_) {
    v = lag_emb[(size_t)lagcat[row] * E_ + (c - RE_ - E_)];
  } else {
    v = el_emb[(size_t)elcat[row] * E_ + (c - RE_ - 2 * E_)];
  }
  out[i] = f2bf(v);
}

// ---------------------------------------------------------------------------
extern "C" void kernel_launch(void* const* d_in, const int* in_sizes, int n_in,
                              void* d_out, int out_size, void* d_ws, size_t ws_size,
                              hipStream_t stream) {
  (void)in_sizes; (void)n_in; (void)out_size; (void)ws_size;
  const int*   enc_ids = (const int*)  d_in[0];
  const int*   cat_ids = (const int*)  d_in[1];
  const int*   resp_ids= (const int*)  d_in[2];
  const int*   ts      = (const int*)  d_in[3];
  const float* el      = (const float*)d_in[4];
  const float* id_emb  = (const float*)d_in[5];
  const float* cat_emb = (const float*)d_in[6];
  const float* enc_lw  = (const float*)d_in[7];
  const float* enc_lb  = (const float*)d_in[8];
  const float* enc_lng = (const float*)d_in[9];
  const float* enc_lnb = (const float*)d_in[10];
  const float* resp_emb= (const float*)d_in[11];
  const float* bn_g    = (const float*)d_in[12];
  const float* bn_b    = (const float*)d_in[13];
  const float* num_w   = (const float*)d_in[14];
  const float* num_b   = (const float*)d_in[15];
  const float* el_emb  = (const float*)d_in[16];
  const float* lag_emb = (const float*)d_in[17];
  const float* dec_lw  = (const float*)d_in[18];
  const float* dec_lb  = (const float*)d_in[19];
  const float* dec_lng = (const float*)d_in[20];
  const float* dec_lnb = (const float*)d_in[21];
  const float* pb_enc  = (const float*)d_in[22];
  const float* pb_dec  = (const float*)d_in[23];
  const float* e_aw    = (const float*)d_in[24];
  const float* e_ab    = (const float*)d_in[25];
  const float* e_ow    = (const float*)d_in[26];
  const float* e_ob    = (const float*)d_in[27];
  const float* e_alg   = (const float*)d_in[28];
  const float* e_alb   = (const float*)d_in[29];
  const float* e_f1w   = (const float*)d_in[30];
  const float* e_f1b   = (const float*)d_in[31];
  const float* e_f2w   = (const float*)d_in[32];
  const float* e_f2b   = (const float*)d_in[33];
  const float* e_flg   = (const float*)d_in[34];
  const float* e_flb   = (const float*)d_in[35];
  const float* d_aw    = (const float*)d_in[36];
  const float* d_ab    = (const float*)d_in[37];
  const float* d_ow    = (const float*)d_in[38];
  const float* d_ob    = (const float*)d_in[39];
  const float* d_alg   = (const float*)d_in[40];
  const float* d_alb   = (const float*)d_in[41];
  const float* d_caw   = (const float*)d_in[42];
  const float* d_cab   = (const float*)d_in[43];
  const float* d_cow   = (const float*)d_in[44];
  const float* d_cob   = (const float*)d_in[45];
  const float* d_clg   = (const float*)d_in[46];
  const float* d_clb   = (const float*)d_in[47];
  const float* d_f1w   = (const float*)d_in[48];
  const float* d_f1b   = (const float*)d_in[49];
  const float* d_f2w   = (const float*)d_in[50];
  const float* d_f2b   = (const float*)d_in[51];
  const float* d_flg   = (const float*)d_in[52];
  const float* d_flb   = (const float*)d_in[53];

  // ---- workspace carving ----
  char* base = (char*)d_ws;
  size_t off = 0;
  auto carve = [&](size_t bytes) -> void* {
    void* p = base + off;
    off += (bytes + 255) & ~(size_t)255;
    return p;
  };
  // convert + swizzle one [K][N] weight matrix into fragment-order tiles
  auto swzw = [&](const float* src, int K, int N) -> const unsigned short* {
    unsigned int* dst = (unsigned int*)carve((size_t)K * N * 2);
    size_t total = (size_t)K * N / 2;
    swz_w_bf16<<<dim3((unsigned)((total + 255) / 256)), 256, 0, stream>>>(src, dst, K, N);
    return (const unsigned short*)dst;
  };
  const size_t DD = (size_t)D_ * D_;

  const unsigned short* sw_enc_lw = swzw(enc_lw, ENC_K, D_);
  const unsigned short* sw_dec_lw = swzw(dec_lw, DEC_K, D_);
  const unsigned short *sw_eq[L_], *sw_ek[L_], *sw_ev[L_], *sw_eo[L_], *sw_e1[L_], *sw_e2[L_];
  const unsigned short *sw_dq[L_], *sw_dk[L_], *sw_dv[L_], *sw_do[L_];
  const unsigned short *sw_cq[L_], *sw_ck[L_], *sw_cv[L_], *sw_co[L_], *sw_d1[L_], *sw_d2[L_];
  for (int i = 0; i < L_; ++i) {
    sw_eq[i] = swzw(e_aw + ((size_t)i * 3 + 0) * DD, D_, D_);
    sw_ek[i] = swzw(e_aw + ((size_t)i * 3 + 1) * DD, D_, D_);
    sw_ev[i] = swzw(e_aw + ((size_t)i * 3 + 2) * DD, D_, D_);
    sw_eo[i] = swzw(e_ow + (size_t)i * DD, D_, D_);
    sw_e1[i] = swzw(e_f1w + (size_t)i * D_ * F_, D_, F_);
    sw_e2[i] = swzw(e_f2w + (size_t)i * F_ * D_, F_, D_);
    sw_dq[i] = swzw(d_aw + ((size_t)i * 3 + 0) * DD, D_, D_);
    sw_dk[i] = swzw(d_aw + ((size_t)i * 3 + 1) * DD, D_, D_);
    sw_dv[i] = swzw(d_aw + ((size_t)i * 3 + 2) * DD, D_, D_);
    sw_do[i] = swzw(d_ow + (size_t)i * DD, D_, D_);
    sw_cq[i] = swzw(d_caw + ((size_t)i * 3 + 0) * DD, D_, D_);
    sw_ck[i] = swzw(d_caw + ((size_t)i * 3 + 1) * DD, D_, D_);
    sw_cv[i] = swzw(d_caw + ((size_t)i * 3 + 2) * DD, D_, D_);
    sw_co[i] = swzw(d_cow + (size_t)i * DD, D_, D_);
    sw_d1[i] = swzw(d_f1w + (size_t)i * D_ * F_, D_, F_);
    sw_d2[i] = swzw(d_f2w + (size_t)i * F_ * D_, F_, D_);
  }

  unsigned short* catb  = (unsigned short*)carve((size_t)MROWS * DEC_K * 2); // reused enc/dec
  float*          tmpf  = (float*)carve((size_t)MROWS * D_ * 4);
  unsigned short* X     = (unsigned short*)carve((size_t)MROWS * F_ * 2);    // qkv/ctx OR ffn hidden
  unsigned short* qb    = X;
  unsigned short* kbuf  = X + (size_t)MROWS * D_;
  unsigned short* vbuf  = X + (size_t)2 * MROWS * D_;
  unsigned short* ctxb  = X + (size_t)3 * MROWS * D_;
  float*          hencf = (float*)carve((size_t)MROWS * D_ * 4);
  unsigned short* hencb = (unsigned short*)carve((size_t)MROWS * D_ * 2);
  float*          hf    = (float*)carve((size_t)MROWS * D_ * 4);
  unsigned short* hb    = (unsigned short*)carve((size_t)MROWS * D_ * 2);
  int*   tsnew  = (int*)carve(MROWS * 4);
  float* n0a    = (float*)carve(MROWS * 4);
  float* n1a    = (float*)carve(MROWS * 4);
  int*   lagcat = (int*)carve(MROWS * 4);
  int*   elcat  = (int*)carve(MROWS * 4);

  auto gemm = [&](const unsigned short* A, int lda, const unsigned short* Wsw,
                  const float* bias, const float* resid, float* oF, unsigned short* oB,
                  int M, int N, int K, int act) {
    dim3 grid(N / 64, M / 128);
    gemm_bf16<<<grid, 256, 0, stream>>>(A, lda, Wsw, bias, resid, oF, oB, M, N, K, act);
  };
  auto attn = [&](const unsigned short* q, const unsigned short* k, const unsigned short* v,
                  const float* pbias, const int* tq, const int* tk,
                  unsigned short* c, int causal) {
    dim3 grid(S_ / 64, H_, B_);
    attn_kernel<<<grid, 128, 0, stream>>>(q, k, v, pbias, tq, tk, c, causal);
  };
  auto ln = [&](const float* x, const float* g, const float* b, float* oF, unsigned short* oB) {
    ln_kernel<<<MROWS, 256, 0, stream>>>(x, g, b, oF, oB);
  };

  // ---- prep + embeddings ----
  prep_kernel<<<MROWS / 256, 256, 0, stream>>>(ts, el, bn_g, bn_b, tsnew, n0a, n1a, lagcat, elcat);

  enc_gather<<<(unsigned)(((size_t)MROWS * ENC_K) / 256), 256, 0, stream>>>(
      enc_ids, cat_ids, id_emb, cat_emb, catb);
  gemm(catb, ENC_K, sw_enc_lw, enc_lb, nullptr, tmpf, nullptr, MROWS, D_, ENC_K, 0);
  ln(tmpf, enc_lng, enc_lnb, hencf, hencb);

  dec_gather<<<(unsigned)(((size_t)MROWS * DEC_K) / 256), 256, 0, stream>>>(
      resp_ids, resp_emb, num_w, num_b, lag_emb, el_emb, n0a, n1a, lagcat, elcat, catb);
  gemm(catb, DEC_K, sw_dec_lw, dec_lb, nullptr, tmpf, nullptr, MROWS, D_, DEC_K, 0);
  ln(tmpf, dec_lng, dec_lnb, hf, hb);

  // ---- encoder layers ----
  for (int i = 0; i < L_; ++i) {
    gemm(hencb, D_, sw_eq[i], e_ab + ((size_t)i * 3 + 0) * D_, nullptr, nullptr, qb,   MROWS, D_, D_, 0);
    gemm(hencb, D_, sw_ek[i], e_ab + ((size_t)i * 3 + 1) * D_, nullptr, nullptr, kbuf, MROWS, D_, D_, 0);
    gemm(hencb, D_, sw_ev[i], e_ab + ((size_t)i * 3 + 2) * D_, nullptr, nullptr, vbuf, MROWS, D_, D_, 0);
    attn(qb, kbuf, vbuf, pb_enc, ts, ts, ctxb, 0);
    gemm(ctxb, D_, sw_eo[i], e_ob + (size_t)i * D_, hencf, tmpf, nullptr, MROWS, D_, D_, 0);
    ln(tmpf, e_alg + (size_t)i * D_, e_alb + (size_t)i * D_, hencf, hencb);
    gemm(hencb, D_, sw_e1[i], e_f1b + (size_t)i * F_, nullptr, nullptr, X, MROWS, F_, D_, 1);
    gemm(X, F_, sw_e2[i], e_f2b + (size_t)i * D_, hencf, tmpf, nullptr, MROWS, D_, F_, 0);
    ln(tmpf, e_flg + (size_t)i * D_, e_flb + (size_t)i * D_, hencf, hencb);
  }

  // ---- decoder layers ----
  for (int i = 0; i < L_; ++i) {
    // self-attention (causal, ts_new scale)
    gemm(hb, D_, sw_dq[i], d_ab + ((size_t)i * 3 + 0) * D_, nullptr, nullptr, qb,   MROWS, D_, D_, 0);
    gemm(hb, D_, sw_dk[i], d_ab + ((size_t)i * 3 + 1) * D_, nullptr, nullptr, kbuf, MROWS, D_, D_, 0);
    gemm(hb, D_, sw_dv[i], d_ab + ((size_t)i * 3 + 2) * D_, nullptr, nullptr, vbuf, MROWS, D_, D_, 0);
    attn(qb, kbuf, vbuf, pb_dec, tsnew, tsnew, ctxb, 1);
    gemm(ctxb, D_, sw_do[i], d_ob + (size_t)i * D_, hf, tmpf, nullptr, MROWS, D_, D_, 0);
    ln(tmpf, d_alg + (size_t)i * D_, d_alb + (size_t)i * D_, hf, hb);

    // cross-attention (q from decoder stream, k/v from encoder output)
    gemm(hb,    D_, sw_cq[i], d_cab + ((size_t)i * 3 + 0) * D_, nullptr, nullptr, qb,   MROWS, D_, D_, 0);
    gemm(hencb, D_, sw_ck[i], d_cab + ((size_t)i * 3 + 1) * D_, nullptr, nullptr, kbuf, MROWS, D_, D_, 0);
    gemm(hencb, D_, sw_cv[i], d_cab + ((size_t)i * 3 + 2) * D_, nullptr, nullptr, vbuf, MROWS, D_, D_, 0);
    attn(qb, kbuf, vbuf, pb_dec, tsnew, ts, ctxb, 0);
    gemm(ctxb, D_, sw_co[i], d_cob + (size_t)i * D_, hf, tmpf, nullptr, MROWS, D_, D_, 0);
    ln(tmpf, d_clg + (size_t)i * D_, d_clb + (size_t)i * D_, hf, hb);

    // FFN
    gemm(hb, D_, sw_d1[i], d_f1b + (size_t)i * F_, nullptr, nullptr, X, MROWS, F_, D_, 1);
    gemm(X, F_, sw_d2[i], d_f2b + (size_t)i * D_, hf, tmpf, nullptr, MROWS, D_, F_, 0);
    ln(tmpf, d_flg + (size_t)i * D_, d_flb + (size_t)i * D_, hf, hb);
  }

  hipMemcpyAsync(d_out, hf, (size_t)MROWS * D_ * 4, hipMemcpyDeviceToDevice, stream);
}
